// LSTMModel_62440234549387
// MI455X (gfx1250) — matrix operand order repaired and checked by reading, not compiled, not run on hardware
//
#include <hip/hip_runtime.h>
#include <hip/hip_bf16.h>

#define B_  256
#define S_  512
#define I_  64
#define H_  512
#define G4_ (4 * H_)

typedef __bf16 bf16;
typedef __attribute__((ext_vector_type(16))) __bf16 v16bf;
typedef __attribute__((ext_vector_type(8)))  __bf16 v8bf;
typedef __attribute__((ext_vector_type(8)))  float  v8f;

union V16U { v16bf v; v8bf h[2]; };

static __device__ __forceinline__ v16bf joinv(v8bf lo, v8bf hi) {
    V16U u; u.h[0] = lo; u.h[1] = hi; return u.v;
}

// Opaque zero: value is 0 at runtime, but the compiler cannot prove it, so
// weight loads are not loop-invariant across timesteps -> no LICM hoist+spill.
static __device__ __forceinline__ unsigned opaque_zero() {
    unsigned z = 0;
    asm volatile("" : "+s"(z));
    return z;
}

// A-matrix 16x32 bf16 fragment (row-major source).
// Lane half=0 holds K {0..7, 16..23}, half=1 holds K {8..15, 24..31}; row M = lane%16.
static __device__ __forceinline__ v16bf load_a_frag(const bf16* row_k0, int half) {
    const bf16* p = row_k0 + half * 8;
    v8bf lo = *(const v8bf*)p;
    v8bf hi = *(const v8bf*)(p + 16);
    return joinv(lo, hi);
}

// B-matrix 32x16 bf16 fragment; weights stored [N,K] row-major.
// Lane holds column N = lane%16, K {0..7, 16..23} for half=0, {8..15, 24..31} for half=1.
static __device__ __forceinline__ v16bf load_b_frag(const bf16* nrow_k0, int half) {
    const bf16* p = nrow_k0 + half * 8;
    v8bf lo = *(const v8bf*)p;
    v8bf hi = *(const v8bf*)(p + 16);
    return joinv(lo, hi);
}

static __device__ __forceinline__ float sigmoidf_(float x) {
    return 1.0f / (1.0f + __expf(-x));
}
static __device__ __forceinline__ float eluf_(float x) {
    return x > 0.0f ? x : (__expf(x) - 1.0f);
}

// Grid-wide split barrier: release-add + acquire-spin at agent scope.
// All blocks are co-resident (64 blocks of 256 threads), so this cannot deadlock.
static __device__ __forceinline__ void grid_sync(unsigned* ctr, unsigned target) {
    __syncthreads();
    if (threadIdx.x == 0) {
        __hip_atomic_fetch_add(ctr, 1u, __ATOMIC_ACQ_REL, __HIP_MEMORY_SCOPE_AGENT);
        while (__hip_atomic_load(ctr, __ATOMIC_ACQUIRE, __HIP_MEMORY_SCOPE_AGENT) < target) {
            __builtin_amdgcn_s_sleep(1);
        }
    }
    __syncthreads();
}

// Persistent LSTM layer kernel: all S timesteps inside one launch.
//   gates = bias + x_s @ wih^T + h_{s-1} @ whh^T   (bf16 WMMA, f32 accum)
// 64 blocks x 256 threads = 512 waves = (B/16)*(H/16) tiles; each wave owns one
// 16x16 (b,h) tile for the whole sequence, keeping cell state c in registers.
// The K loop is software-pipelined 1 iteration deep: the next iteration's
// A-fragment and all 4 gate B-fragments are in flight while the current
// iteration's 4 WMMAs execute.
template <int KIN>
__global__ __launch_bounds__(256) void lstm_layer_kernel(
    const bf16* __restrict__ xseq,   // [S, B, KIN]
    const bf16* __restrict__ h0,     // [B, H] zeros
    const bf16* __restrict__ wih,    // [4H, KIN] row-major (N,K)
    const bf16* __restrict__ whh,    // [4H, H]   row-major (N,K)
    const float* __restrict__ bias,  // [4H]
    bf16* __restrict__ hseq,         // [S, B, H] out
    unsigned* __restrict__ ctr)      // grid barrier counter (pre-zeroed)
{
    const int lane   = threadIdx.x & 31;
    const int wave   = (blockIdx.x * blockDim.x + threadIdx.x) >> 5;
    const int nTiles = H_ / 16;                 // 32
    const int mTile  = wave / nTiles;
    const int nTile  = wave % nTiles;
    const int half   = lane >> 4;
    const int lane16 = lane & 15;

    const int mRow = mTile * 16 + lane16;       // A-matrix row for this lane
    const int nCol = nTile * 16 + lane16;       // weight row (output column)

    // Per-layer constants: biases loaded once.
    const float bi = bias[0 * H_ + nCol];
    const float bf = bias[1 * H_ + nCol];
    const float bg = bias[2 * H_ + nCol];
    const float bo = bias[3 * H_ + nCol];

    const bf16* wihRow[4];
    const bf16* whhRow[4];
    #pragma unroll
    for (int g = 0; g < 4; ++g) {
        wihRow[g] = wih + (size_t)(g * H_ + nCol) * KIN;
        whhRow[g] = whh + (size_t)(g * H_ + nCol) * H_;
    }

    // Cell state lives in registers for the whole sequence.
    float c[8];
    #pragma unroll
    for (int r = 0; r < 8; ++r) c[r] = 0.0f;

    const unsigned nb = (unsigned)gridDim.x;
    const bf16* hprev = h0 + (size_t)mRow * H_;

    constexpr int ITER = (KIN + H_) / 32;   // unified K space: x part then h part

    for (int s = 0; s < S_; ++s) {
        const unsigned z = opaque_zero();   // anti-LICM for weight loads

        const bf16* arow = xseq + (size_t)s * B_ * KIN + (size_t)mRow * KIN;

        // Load one unified iteration's fragments (A + 4 gate B-frags).
        auto load_iter = [&](int it, v16bf& a, v16bf b[4]) {
            const int kk = it * 32;
            if (kk < KIN) {
                a = load_a_frag(arow + kk, half);
                #pragma unroll
                for (int g = 0; g < 4; ++g)
                    b[g] = load_b_frag(wihRow[g] + kk + z, half);
            } else {
                const int k2 = kk - KIN;
                a = load_a_frag(hprev + k2, half);
                #pragma unroll
                for (int g = 0; g < 4; ++g)
                    b[g] = load_b_frag(whhRow[g] + k2 + z, half);
            }
        };

        v8f acc[4];
        #pragma unroll
        for (int g = 0; g < 4; ++g) acc[g] = (v8f)(0.0f);

        v16bf aC, bC[4];
        load_iter(0, aC, bC);

        #pragma unroll
        for (int it = 0; it < ITER; ++it) {
            v16bf aN, bN[4];
            if (it + 1 < ITER) load_iter(it + 1, aN, bN);  // prefetch next iter

            #pragma unroll
            for (int g = 0; g < 4; ++g) {
                acc[g] = __builtin_amdgcn_wmma_f32_16x16x32_bf16(
                    false, aC, false, bC[g], (short)0, acc[g], false, false);
            }

            if (it + 1 < ITER) {
                aC = aN;
                #pragma unroll
                for (int g = 0; g < 4; ++g) bC[g] = bN[g];
            }
        }

        // ---- cell update (custom elu activations), c stays in registers ----
        // C/D layout: VGPR r holds element (M = r + 8*half, N = lane16).
        bf16* hout = hseq + (size_t)s * B_ * H_;
        #pragma unroll
        for (int r = 0; r < 8; ++r) {
            const int m = mTile * 16 + r + 8 * half;
            float gi = sigmoidf_(acc[0][r] + bi);
            float gf = sigmoidf_(acc[1][r] + bf);
            float gg = eluf_    (acc[2][r] + bg);
            float go = sigmoidf_(acc[3][r] + bo);
            c[r] = gf * c[r] + gi * gg;
            hout[(size_t)m * H_ + nCol] = (bf16)(go * eluf_(c[r]));
        }

        hprev = hseq + (size_t)s * B_ * H_ + (size_t)mRow * H_;

        // Make h_s visible to every block before anyone starts step s+1.
        grid_sync(ctr, nb * (unsigned)(s + 1));
    }
}

// ---- helpers: conversions / init / epilogue ----

__global__ __launch_bounds__(256) void convert_f32_to_bf16(
    const float* __restrict__ src, bf16* __restrict__ dst, int n)
{
    int i = blockIdx.x * blockDim.x + threadIdx.x;
    if (i < n) dst[i] = (bf16)src[i];
}

// x: [B, S, I] f32 -> [S, B, I] bf16 (seq-first, step-contiguous slices)
__global__ __launch_bounds__(256) void convert_x_kernel(
    const float* __restrict__ src, bf16* __restrict__ dst)
{
    int i = blockIdx.x * blockDim.x + threadIdx.x;
    int n = B_ * S_ * I_;
    if (i >= n) return;
    int b = i / (S_ * I_);
    int r = i - b * (S_ * I_);
    int s = r / I_;
    int col = r - s * I_;
    dst[(size_t)s * B_ * I_ + (size_t)b * I_ + col] = (bf16)src[i];
}

__global__ __launch_bounds__(256) void init_kernel(
    bf16* __restrict__ h0, unsigned* __restrict__ ctrs)
{
    int i = blockIdx.x * blockDim.x + threadIdx.x;
    if (i < B_ * H_) h0[i] = (bf16)0.0f;
    if (i < 3) ctrs[i] = 0u;
}

// out[b] = sum_h hlast[b,h] * fc_w[h] + fc_b
__global__ __launch_bounds__(256) void fc_kernel(
    const bf16* __restrict__ hlast, const float* __restrict__ w,
    const float* __restrict__ bias, float* __restrict__ out)
{
    int b = blockIdx.x * blockDim.x + threadIdx.x;
    if (b >= B_) return;
    float s = 0.0f;
    const bf16* row = hlast + (size_t)b * H_;
    #pragma unroll 8
    for (int h = 0; h < H_; ++h) s += (float)row[h] * w[h];
    out[b] = s + bias[0];
}

extern "C" void kernel_launch(void* const* d_in, const int* in_sizes, int n_in,
                              void* d_out, int out_size, void* d_ws, size_t ws_size,
                              hipStream_t stream) {
    (void)in_sizes; (void)n_in; (void)out_size; (void)ws_size;

    const float* x    = (const float*)d_in[0];
    const float* wih1 = (const float*)d_in[1];
    const float* whh1 = (const float*)d_in[2];
    const float* b1   = (const float*)d_in[3];
    const float* wih2 = (const float*)d_in[4];
    const float* whh2 = (const float*)d_in[5];
    const float* b2   = (const float*)d_in[6];
    const float* wih3 = (const float*)d_in[7];
    const float* whh3 = (const float*)d_in[8];
    const float* b3   = (const float*)d_in[9];
    const float* fcw  = (const float*)d_in[10];
    const float* fcb  = (const float*)d_in[11];

    char* ws = (char*)d_ws;
    size_t off = 0;
    auto take = [&](size_t bytes) -> char* {
        char* p = ws + off;
        off += (bytes + 255) & ~(size_t)255;
        return p;
    };

    bf16* wih1b = (bf16*)take((size_t)G4_ * I_ * sizeof(bf16));
    bf16* whh1b = (bf16*)take((size_t)G4_ * H_ * sizeof(bf16));
    bf16* wih2b = (bf16*)take((size_t)G4_ * H_ * sizeof(bf16));
    bf16* whh2b = (bf16*)take((size_t)G4_ * H_ * sizeof(bf16));
    bf16* wih3b = (bf16*)take((size_t)G4_ * H_ * sizeof(bf16));
    bf16* whh3b = (bf16*)take((size_t)G4_ * H_ * sizeof(bf16));
    bf16* xb    = (bf16*)take((size_t)S_ * B_ * I_ * sizeof(bf16));
    bf16* bufA  = (bf16*)take((size_t)S_ * B_ * H_ * sizeof(bf16)); // L1 out, L3 out
    bf16* bufB  = (bf16*)take((size_t)S_ * B_ * H_ * sizeof(bf16)); // L2 out
    bf16* h0    = (bf16*)take((size_t)B_ * H_ * sizeof(bf16));
    unsigned* ctrs = (unsigned*)take(3 * sizeof(unsigned));

    // ---- weight / input conversion to bf16, barrier counters + h0 init ----
    {
        int n = G4_ * I_;
        convert_f32_to_bf16<<<(n + 255) / 256, 256, 0, stream>>>(wih1, wih1b, n);
        n = G4_ * H_;
        convert_f32_to_bf16<<<(n + 255) / 256, 256, 0, stream>>>(whh1, whh1b, n);
        convert_f32_to_bf16<<<(n + 255) / 256, 256, 0, stream>>>(wih2, wih2b, n);
        convert_f32_to_bf16<<<(n + 255) / 256, 256, 0, stream>>>(whh2, whh2b, n);
        convert_f32_to_bf16<<<(n + 255) / 256, 256, 0, stream>>>(wih3, wih3b, n);
        convert_f32_to_bf16<<<(n + 255) / 256, 256, 0, stream>>>(whh3, whh3b, n);
        int nx = B_ * S_ * I_;
        convert_x_kernel<<<(nx + 255) / 256, 256, 0, stream>>>(x, xb);
        init_kernel<<<(B_ * H_ + 255) / 256, 256, 0, stream>>>(h0, ctrs);
    }

    const int layerBlocks = (B_ / 16) * (H_ / 16) * 32 / 256; // 512 waves -> 64 blocks

    // ---- three persistent layer kernels ----
    lstm_layer_kernel<I_><<<layerBlocks, 256, 0, stream>>>(
        xb, h0, wih1b, whh1b, b1, bufA, &ctrs[0]);
    lstm_layer_kernel<H_><<<layerBlocks, 256, 0, stream>>>(
        bufA, h0, wih2b, whh2b, b2, bufB, &ctrs[1]);
    lstm_layer_kernel<H_><<<layerBlocks, 256, 0, stream>>>(
        bufB, h0, wih3b, whh3b, b3, bufA, &ctrs[2]);

    // ---- final FC on last timestep of layer 3 ----
    fc_kernel<<<(B_ + 255) / 256, 256, 0, stream>>>(
        bufA + (size_t)(S_ - 1) * B_ * H_, fcw, fcb, (float*)d_out);
}
